// LearnedEdge_79267916415039
// MI455X (gfx1250) — matrix operand order repaired and checked by reading, not compiled, not run on hardware
//
#include <hip/hip_runtime.h>
#include <hip/hip_bf16.h>

typedef __attribute__((ext_vector_type(16))) _Float16 v16h;
typedef __attribute__((ext_vector_type(8)))  _Float16 v8h;
typedef __attribute__((ext_vector_type(8)))  float    v8f;

#define BB 16
#define NN 2048
#define DD 128
#define SS 5

__device__ inline v16h mk16(v8h lo, v8h hi) {
  v16h r;
#pragma unroll
  for (int i = 0; i < 8; i++) { r[i] = lo[i]; r[i + 8] = hi[i]; }
  return r;
}

// Pre-kernel: c1[b][k] = b1[k] + curr_b @ W1[0:128]; transpose+convert W1[128:256] and W2 to f16.
__global__ void __launch_bounds__(256) le_prep(const float* __restrict__ nodes,
                                               const int* __restrict__ num_nodes,
                                               const float* __restrict__ W1,
                                               const float* __restrict__ b1,
                                               const float* __restrict__ W2,
                                               float* __restrict__ c1,
                                               _Float16* __restrict__ wt1,
                                               _Float16* __restrict__ wt2) {
  int t = blockIdx.x * 256 + threadIdx.x;   // 0..16383
  if (t < BB * DD) {
    int b = t >> 7, k = t & 127;
    int nn = num_nodes[b];
    const float* cur = nodes + ((size_t)b * NN + nn) * DD;
    float s = b1[k];
#pragma unroll 4
    for (int j = 0; j < DD; j++) s += cur[j] * W1[j * DD + k];
    c1[t] = s;
  }
  // transpose: wt[n*128+k] = W[k][n]
  int n = t & 127, k = t >> 7;              // k in 0..127
  wt1[n * DD + k] = (_Float16)W1[(DD + k) * DD + n];
  wt2[n * DD + k] = (_Float16)W2[k * DD + n];
}

// MLP kernel: one wave32 per 16-row tile; f16 WMMA 16x16x32, f32 accumulate.
__global__ void __launch_bounds__(256) le_mlp(const float* __restrict__ nodes,
                                              const float* __restrict__ c1,
                                              const _Float16* __restrict__ wt1,
                                              const _Float16* __restrict__ wt2,
                                              const float* __restrict__ b2,
                                              const float* __restrict__ g1,
                                              const float* __restrict__ be1,
                                              const float* __restrict__ g2,
                                              const float* __restrict__ be2,
                                              const float* __restrict__ W3,
                                              const float* __restrict__ b3,
                                              float* __restrict__ lg) {
  __shared__ _Float16 hbuf[8 * 16 * DD];    // 4 KB per wave, 32 KB total
  const int tid  = threadIdx.x;
  const int wave = tid >> 5;
  const int lane = tid & 31;
  const int M    = lane & 15;               // row within tile (A), col within N-tile (B/C)
  const int g    = lane >> 4;               // half-wave group
  const int tile = blockIdx.x * 8 + wave;   // 0..2047
  const int b    = tile >> 7;
  const int n0   = (tile & 127) << 4;
  _Float16* hb = hbuf + wave * (16 * DD);

  // stage node tile as f16 (A source for layer 1)
  for (int idx = lane; idx < 16 * DD; idx += 32) {
    int r = idx >> 7, c = idx & 127;
    hb[idx] = (_Float16)nodes[((size_t)b * NN + n0 + r) * DD + c];
  }
  __syncthreads();

  // ---- layer 1: h1 = relu(node @ W1b + c1[b]) ----
  v16h a[4];
#pragma unroll
  for (int kk = 0; kk < 4; kk++) {
    const v8h* lo = (const v8h*)(hb + M * DD + kk * 32 + 8 * g);
    const v8h* hi = (const v8h*)(hb + M * DD + kk * 32 + 16 + 8 * g);
    a[kk] = mk16(*lo, *hi);
  }
  __syncthreads();

#pragma unroll
  for (int ct = 0; ct < 8; ct++) {
    int col = ct * 16 + M;
    v8f acc = {0.f, 0.f, 0.f, 0.f, 0.f, 0.f, 0.f, 0.f};
#pragma unroll
    for (int kk = 0; kk < 4; kk++) {
      const v8h* bp = (const v8h*)(wt1 + (size_t)col * DD + kk * 32 + 8 * g);
      v16h bf = mk16(bp[0], bp[2]);
      acc = __builtin_amdgcn_wmma_f32_16x16x32_f16(false, a[kk], false, bf,
                                                   (short)0, acc, false, false);
    }
    float cc = c1[b * DD + col];
#pragma unroll
    for (int i = 0; i < 8; i++) {
      float v = acc[i] + cc;
      hb[(i + 8 * g) * DD + col] = (_Float16)(v > 0.f ? v : 0.f);
    }
  }
  __syncthreads();

  // ---- LayerNorm 1 (one row per lane, lanes 0..15) ----
  if (lane < 16) {
    _Float16* row = hb + lane * DD;
    float s = 0.f;
    for (int k = 0; k < DD; k++) s += (float)row[k];
    float m = s * (1.f / DD);
    float v2 = 0.f;
    for (int k = 0; k < DD; k++) { float d = (float)row[k] - m; v2 += d * d; }
    float r = rsqrtf(v2 * (1.f / DD) + 1e-5f);
    for (int k = 0; k < DD; k++)
      row[k] = (_Float16)(((float)row[k] - m) * r * g1[k] + be1[k]);
  }
  __syncthreads();

  // ---- layer 2: h2 = relu(h1n @ W2 + b2) ----
#pragma unroll
  for (int kk = 0; kk < 4; kk++) {
    const v8h* lo = (const v8h*)(hb + M * DD + kk * 32 + 8 * g);
    const v8h* hi = (const v8h*)(hb + M * DD + kk * 32 + 16 + 8 * g);
    a[kk] = mk16(*lo, *hi);
  }
  __syncthreads();

#pragma unroll
  for (int ct = 0; ct < 8; ct++) {
    int col = ct * 16 + M;
    v8f acc = {0.f, 0.f, 0.f, 0.f, 0.f, 0.f, 0.f, 0.f};
#pragma unroll
    for (int kk = 0; kk < 4; kk++) {
      const v8h* bp = (const v8h*)(wt2 + (size_t)col * DD + kk * 32 + 8 * g);
      v16h bf = mk16(bp[0], bp[2]);
      acc = __builtin_amdgcn_wmma_f32_16x16x32_f16(false, a[kk], false, bf,
                                                   (short)0, acc, false, false);
    }
    float bb = b2[col];
#pragma unroll
    for (int i = 0; i < 8; i++) {
      float v = acc[i] + bb;
      hb[(i + 8 * g) * DD + col] = (_Float16)(v > 0.f ? v : 0.f);
    }
  }
  __syncthreads();

  // ---- LayerNorm 2 ----
  if (lane < 16) {
    _Float16* row = hb + lane * DD;
    float s = 0.f;
    for (int k = 0; k < DD; k++) s += (float)row[k];
    float m = s * (1.f / DD);
    float v2 = 0.f;
    for (int k = 0; k < DD; k++) { float d = (float)row[k] - m; v2 += d * d; }
    float r = rsqrtf(v2 * (1.f / DD) + 1e-5f);
    for (int k = 0; k < DD; k++)
      row[k] = (_Float16)(((float)row[k] - m) * r * g2[k] + be2[k]);
  }
  __syncthreads();

  // ---- layer 3: logits = h2n @ W3 + b3 ----
  if (lane < 16) {
    _Float16* row = hb + lane * DD;
    float s = b3[0];
    for (int k = 0; k < DD; k++) s += (float)row[k] * W3[k];
    lg[(size_t)b * NN + n0 + lane] = s;
  }
}

// Argmax over masked logits + gumbel per sample, then scatter the new adj row.
__global__ void __launch_bounds__(256) le_pick(const float* __restrict__ lg,
                                               const float* __restrict__ gumbel,
                                               const int* __restrict__ num_nodes,
                                               const float* __restrict__ adj,
                                               float* __restrict__ out_adj) {
  __shared__ float sv[256];
  __shared__ int   si[256];
  __shared__ int   win[SS];
  const int b = blockIdx.x;
  const int tid = threadIdx.x;
  const int nn = num_nodes[b];
  const float* lp = lg + (size_t)b * NN;

  for (int s = 0; s < SS; s++) {
    const float* gp = gumbel + ((size_t)s * BB + b) * NN;
    float bv = -3.4e38f; int bi = 0;
    for (int j = tid; j < NN; j += 256) {
      float v = ((j < nn) ? lp[j] : -1e10f) + gp[j];
      if (v > bv || (v == bv && j < bi)) { bv = v; bi = j; }
    }
    sv[tid] = bv; si[tid] = bi;
    __syncthreads();
    for (int off = 128; off > 0; off >>= 1) {
      if (tid < off) {
        float ov = sv[tid + off]; int oi = si[tid + off];
        if (ov > sv[tid] || (ov == sv[tid] && oi < si[tid])) { sv[tid] = ov; si[tid] = oi; }
      }
      __syncthreads();
    }
    if (tid == 0) win[s] = si[0];
    __syncthreads();
  }

  const size_t rowOff = ((size_t)b * NN + nn) * NN;
  for (int j = tid; j < NN; j += 256) {
    float old = adj[rowOff + j];
    int cnt = 0;
#pragma unroll
    for (int s = 0; s < SS; s++) cnt += (win[s] == j) ? 1 : 0;
    float edges = (cnt > 0) ? 1.f : 0.f;
    float nr = ((edges + old) > 0.f) ? 1.f : 0.f;
    out_adj[rowOff + j] = (j < nn) ? nr : old;
  }
}

extern "C" void kernel_launch(void* const* d_in, const int* in_sizes, int n_in,
                              void* d_out, int out_size, void* d_ws, size_t ws_size,
                              hipStream_t stream) {
  (void)in_sizes; (void)n_in; (void)out_size; (void)ws_size;
  const float* nodes     = (const float*)d_in[0];
  const float* adj       = (const float*)d_in[1];
  const float* weights   = (const float*)d_in[2];
  const int*   num_nodes = (const int*)d_in[3];
  const float* W1  = (const float*)d_in[4];
  const float* b1  = (const float*)d_in[5];
  const float* g1  = (const float*)d_in[6];
  const float* be1 = (const float*)d_in[7];
  const float* W2  = (const float*)d_in[8];
  const float* b2  = (const float*)d_in[9];
  const float* g2  = (const float*)d_in[10];
  const float* be2 = (const float*)d_in[11];
  const float* W3  = (const float*)d_in[12];
  const float* b3  = (const float*)d_in[13];
  const float* gumbel = (const float*)d_in[14];

  float* out_adj = (float*)d_out;
  float* out_w   = out_adj + (size_t)BB * NN * NN;
  const size_t mat_bytes = (size_t)BB * NN * NN * sizeof(float);

  // bulk passthrough copies (bandwidth-bound part: ~1 GB total traffic)
  hipMemcpyAsync(out_adj, adj, mat_bytes, hipMemcpyDeviceToDevice, stream);
  hipMemcpyAsync(out_w, weights, mat_bytes, hipMemcpyDeviceToDevice, stream);

  // workspace carve-out: c1 [B*128] f32 | logits [B*N] f32 | wt1,wt2 [128*128] f16 each
  float* c1 = (float*)d_ws;
  float* lg = c1 + BB * DD;
  _Float16* wt1 = (_Float16*)(lg + (size_t)BB * NN);
  _Float16* wt2 = wt1 + DD * DD;

  le_prep<<<64, 256, 0, stream>>>(nodes, num_nodes, W1, b1, W2, c1, wt1, wt2);
  le_mlp<<<256, 256, 0, stream>>>(nodes, c1, wt1, wt2, b2, g1, be1, g2, be2, W3, b3, lg);
  le_pick<<<BB, 256, 0, stream>>>(lg, gumbel, num_nodes, adj, out_adj);
}
